// MultiHeadSelfAttention_54082228191913
// MI455X (gfx1250) — matrix operand order repaired and checked by reading, not compiled, not run on hardware
//
#include <hip/hip_runtime.h>
#include <hip/hip_bf16.h>

typedef __attribute__((ext_vector_type(16))) __bf16 v16bf;
typedef __attribute__((ext_vector_type(8)))  __bf16 v8bf;
typedef __attribute__((ext_vector_type(4)))  __bf16 v4bf;
typedef __attribute__((ext_vector_type(8)))  float  v8f;

#define HDIM 768
#define NHEAD 12
#define HEADD 64
#define BATCH 2
#define SEQ 2048
#define MTOT (BATCH * SEQ)   // 4096

__device__ __forceinline__ v16bf cat8(v8bf lo, v8bf hi) {
    return __builtin_shufflevector(lo, hi, 0,1,2,3,4,5,6,7,8,9,10,11,12,13,14,15);
}

// gfx1250 async copy: global -> LDS, 16B per lane, tracked by ASYNCcnt.
// the instruction offset is added to both the LDS and the global address
__device__ __forceinline__ void async_b128(unsigned lds_off, const void* gaddr) {
    asm volatile("global_load_async_to_lds_b128 %0, %1, off"
                 :: "v"(lds_off), "v"(gaddr) : "memory");
}
__device__ __forceinline__ void async_b128_o16(unsigned lds_off, const void* gaddr) {
    asm volatile("global_load_async_to_lds_b128 %0, %1, off offset:16"
                 :: "v"(lds_off), "v"(gaddr) : "memory");
}
__device__ __forceinline__ void wait_async0() {
    asm volatile("s_wait_asynccnt 0x0" ::: "memory");
}

// ---------------------------------------------------------------------------
// f32 -> bf16 conversion (bandwidth-trivial; everything then lives in L2)
// ---------------------------------------------------------------------------
__global__ void cvt_f32_bf16(const float* __restrict__ in, __bf16* __restrict__ out, int n) {
    int i = (blockIdx.x * blockDim.x + threadIdx.x) * 4;
    if (i >= n) return;
    float4 f = *(const float4*)(in + i);
    v4bf o;
    o[0] = (__bf16)f.x; o[1] = (__bf16)f.y; o[2] = (__bf16)f.z; o[3] = (__bf16)f.w;
    *(v4bf*)(out + i) = o;
}

// ---------------------------------------------------------------------------
// Y[m,n] = sum_k A[m,k] * W[n,k] + bias[n]     (A: MTOT x 768, W: 768 x 768)
// Block 128x128, 8 waves; wave tile 32x64 = 2x4 grid of 16x16 WMMA tiles.
// A/B 128x32 tiles staged in LDS via async b128 copies, double-buffered.
// MODE 0: -> Q  bf16 [B,NH,S,HD], scaled by 1/sqrt(HD)
// MODE 1: -> K  bf16 [B,NH,S,HD]
// MODE 2: -> Vt bf16 [B,NH,HD,S]  (transposed for the P@V WMMA B-fragments)
// MODE 3: -> out f32 [MTOT, H]
// ---------------------------------------------------------------------------
template<int MODE>
__global__ __launch_bounds__(256) void gemm_bf16_wmma(
    const __bf16* __restrict__ A, const __bf16* __restrict__ W,
    const float* __restrict__ bias, void* __restrict__ outp)
{
    // [buf][A=0/B=1][128 rows][32 k] bf16  -> 32 KB total
    __shared__ __bf16 lds[2][2][128 * 32];

    const int tid  = threadIdx.x;
    const int lane = tid & 31;
    const int wave = tid >> 5;
    const int lr = lane & 15;       // row (A) / col (B,C)
    const int lh = lane >> 4;       // half-wave selector
    const int wm = wave >> 1;       // 0..3 (M)
    const int wn = wave & 1;        // 0..1 (N)
    const int m0 = blockIdx.x * 128;
    const int n0 = blockIdx.y * 128;

    // tile-fill assignment: each thread owns 32B (2 x b128) of A-tile and B-tile
    const int frow = tid >> 1;            // 0..127
    const int fcol = (tid & 1) * 16;      // element offset 0 or 16
    const __bf16* ga = A + (size_t)(m0 + frow) * HDIM + fcol;
    const __bf16* gw = W + (size_t)(n0 + frow) * HDIM + fcol;
    const unsigned lbase = (unsigned)(uintptr_t)&lds[0][0][0];
    const unsigned foff  = (unsigned)(frow * 32 + fcol) * 2;   // bytes within a tile

    v8f acc[2][4];
#pragma unroll
    for (int i = 0; i < 2; i++)
#pragma unroll
        for (int j = 0; j < 4; j++)
#pragma unroll
            for (int v = 0; v < 8; v++) acc[i][j][v] = 0.0f;

    auto issue = [&](int buf, int k) {
        const unsigned lA = lbase + (unsigned)(buf * 2 + 0) * (128 * 32 * 2) + foff;
        const unsigned lB = lbase + (unsigned)(buf * 2 + 1) * (128 * 32 * 2) + foff;
        const __bf16* pa = ga + k;
        const __bf16* pw = gw + k;
        async_b128(lA, pa);
        async_b128_o16(lA, pa);
        async_b128(lB, pw);
        async_b128_o16(lB, pw);
    };

    const int NK = HDIM / 32;  // 24
    issue(0, 0);
    for (int it = 0; it < NK; it++) {
        const int p = it & 1;
        wait_async0();           // this wave's fills of buf[p] have landed
        __syncthreads();         // everyone's fills have landed
        if (it + 1 < NK) issue(1 - p, (it + 1) * 32);

        const __bf16* As = &lds[p][0][0];
        const __bf16* Bs = &lds[p][1][0];
        v16bf a[2], b[4];
#pragma unroll
        for (int i = 0; i < 2; i++) {
            const __bf16* pa = As + (wm * 32 + i * 16 + lr) * 32 + lh * 8;
            a[i] = cat8(*(const v8bf*)pa, *(const v8bf*)(pa + 16));
        }
#pragma unroll
        for (int j = 0; j < 4; j++) {
            const __bf16* pb = Bs + (wn * 64 + j * 16 + lr) * 32 + lh * 8;
            b[j] = cat8(*(const v8bf*)pb, *(const v8bf*)(pb + 16));
        }
#pragma unroll
        for (int i = 0; i < 2; i++)
#pragma unroll
            for (int j = 0; j < 4; j++)
                acc[i][j] = __builtin_amdgcn_wmma_f32_16x16x32_bf16(
                    false, a[i], false, b[j], (short)0, acc[i][j], false, false);
        __syncthreads();         // done reading buf[p]; safe to refill next round
    }

    const int wm0 = m0 + wm * 32;
    const int wn0 = n0 + wn * 64;
#pragma unroll
    for (int i = 0; i < 2; i++) {
#pragma unroll
        for (int j = 0; j < 4; j++) {
            const int n = wn0 + j * 16 + lr;
            const float bn = bias[n];
            if (MODE == 0 || MODE == 1) {
                __bf16* out = (__bf16*)outp;
                const int h = n >> 6, d = n & 63;
#pragma unroll
                for (int v = 0; v < 8; v++) {
                    const int m = wm0 + i * 16 + lh * 8 + v;
                    const int bb = m >> 11, s = m & 2047;
                    float val = acc[i][j][v] + bn;
                    if (MODE == 0) val *= 0.125f;   // 1/sqrt(64) folded into Q
                    out[(((size_t)bb * NHEAD + h) * SEQ + s) * HEADD + d] = (__bf16)val;
                }
            } else if (MODE == 2) {
                __bf16* out = (__bf16*)outp;
                const int h = n >> 6, d = n & 63;
                const int mbase = wm0 + i * 16 + lh * 8;
                const int bb = mbase >> 11, sbase = mbase & 2047;
                v8bf pk;
#pragma unroll
                for (int v = 0; v < 8; v++) pk[v] = (__bf16)(acc[i][j][v] + bn);
                *(v8bf*)(out + (((size_t)bb * NHEAD + h) * HEADD + d) * SEQ + sbase) = pk;
            } else {
                float* out = (float*)outp;
#pragma unroll
                for (int v = 0; v < 8; v++) {
                    const int m = wm0 + i * 16 + lh * 8 + v;
                    out[(size_t)m * HDIM + n] = acc[i][j][v] + bn;
                }
            }
        }
    }
}

// ---------------------------------------------------------------------------
// Flash attention per (batch, head, 64-row Q block). 4 waves x 16 Q rows.
// scores = (Q/8) @ K^T  (WMMA),   online softmax,   ctx += P @ V (WMMA).
// P re-laid C->A fragments via per-wave LDS (bf16).
// ---------------------------------------------------------------------------
__global__ __launch_bounds__(128) void flash_attn_wmma(
    const __bf16* __restrict__ Q, const __bf16* __restrict__ K,
    const __bf16* __restrict__ Vt, const int* __restrict__ mask,
    __bf16* __restrict__ ctx)
{
    __shared__ __bf16 plds[4][16 * 64];
    const int lane = threadIdx.x & 31;
    const int wave = threadIdx.x >> 5;
    const int lr = lane & 15, lh = lane >> 4;
    const int h = blockIdx.y, bb = blockIdx.z;
    const int q0 = blockIdx.x * 64 + wave * 16;

    const size_t bh = (size_t)bb * NHEAD + h;
    const __bf16* Qb = Q + bh * SEQ * HEADD;
    const __bf16* Kb = K + bh * SEQ * HEADD;
    const __bf16* Vb = Vt + bh * HEADD * SEQ;
    const int* mk = mask + bb * SEQ;

    // Q A-fragments (held for whole kv loop), d-steps 0 and 32
    v16bf aq[2];
    {
        const __bf16* qr = Qb + (size_t)(q0 + lr) * HEADD;
#pragma unroll
        for (int ds = 0; ds < 2; ds++) {
            const __bf16* p = qr + ds * 32 + lh * 8;
            aq[ds] = cat8(*(const v8bf*)p, *(const v8bf*)(p + 16));
        }
    }

    v8f acc[4];
    float mrow[8], lrow[8];
#pragma unroll
    for (int t = 0; t < 4; t++)
#pragma unroll
        for (int v = 0; v < 8; v++) acc[t][v] = 0.0f;
#pragma unroll
    for (int v = 0; v < 8; v++) { mrow[v] = -1e30f; lrow[v] = 0.0f; }

    for (int kv = 0; kv < SEQ; kv += 64) {
        // ---- scores: 4 tiles of 16x16, 2 WMMAs each (d = 0..31, 32..63) ----
        v8f sc[4];
#pragma unroll
        for (int t = 0; t < 4; t++) {
            const __bf16* kr = Kb + (size_t)(kv + t * 16 + lr) * HEADD;
            v16bf b0 = cat8(*(const v8bf*)(kr + lh * 8), *(const v8bf*)(kr + lh * 8 + 16));
            v16bf b1 = cat8(*(const v8bf*)(kr + 32 + lh * 8), *(const v8bf*)(kr + 32 + lh * 8 + 16));
            v8f z;
#pragma unroll
            for (int v = 0; v < 8; v++) z[v] = 0.0f;
            z = __builtin_amdgcn_wmma_f32_16x16x32_bf16(false, aq[0], false, b0, (short)0, z, false, false);
            z = __builtin_amdgcn_wmma_f32_16x16x32_bf16(false, aq[1], false, b1, (short)0, z, false, false);
            const int mv = mk[kv + t * 16 + lr];   // column mask
            if (mv == 0) {
#pragma unroll
                for (int v = 0; v < 8; v++) z[v] = -1e9f;
            }
            sc[t] = z;
        }

        // ---- online softmax: row max / row sum via half-wave shfl_xor ----
        float tmx[8];
#pragma unroll
        for (int v = 0; v < 8; v++) {
            float t = fmaxf(fmaxf(sc[0][v], sc[1][v]), fmaxf(sc[2][v], sc[3][v]));
            t = fmaxf(t, __shfl_xor(t, 1, 32));
            t = fmaxf(t, __shfl_xor(t, 2, 32));
            t = fmaxf(t, __shfl_xor(t, 4, 32));
            t = fmaxf(t, __shfl_xor(t, 8, 32));
            tmx[v] = t;
        }
        float alpha[8];
#pragma unroll
        for (int v = 0; v < 8; v++) {
            float mn = fmaxf(mrow[v], tmx[v]);
            alpha[v] = __expf(mrow[v] - mn);
            mrow[v] = mn;
        }
        float rs[8];
#pragma unroll
        for (int v = 0; v < 8; v++) rs[v] = 0.0f;
#pragma unroll
        for (int t = 0; t < 4; t++)
#pragma unroll
            for (int v = 0; v < 8; v++) {
                float p = __expf(sc[t][v] - mrow[v]);
                sc[t][v] = p;
                rs[v] += p;
            }
#pragma unroll
        for (int v = 0; v < 8; v++) {
            float t = rs[v];
            t += __shfl_xor(t, 1, 32);
            t += __shfl_xor(t, 2, 32);
            t += __shfl_xor(t, 4, 32);
            t += __shfl_xor(t, 8, 32);
            lrow[v] = lrow[v] * alpha[v] + t;
        }
#pragma unroll
        for (int t = 0; t < 4; t++)
#pragma unroll
            for (int v = 0; v < 8; v++) acc[t][v] *= alpha[v];

        // ---- C-fragment P -> bf16 LDS -> A-fragment (per-wave region) ----
#pragma unroll
        for (int t = 0; t < 4; t++)
#pragma unroll
            for (int v = 0; v < 8; v++)
                plds[wave][(lh * 8 + v) * 64 + t * 16 + lr] = (__bf16)sc[t][v];

        // ---- ctx += P @ V : k-steps 0,32 over the 64 kv columns ----
#pragma unroll
        for (int ks = 0; ks < 2; ks++) {
            const __bf16* pp = &plds[wave][lr * 64 + ks * 32 + lh * 8];
            v16bf pa = cat8(*(const v8bf*)pp, *(const v8bf*)(pp + 16));
#pragma unroll
            for (int t = 0; t < 4; t++) {
                const __bf16* vr = Vb + (size_t)(t * 16 + lr) * SEQ + kv + ks * 32 + lh * 8;
                v16bf bv = cat8(*(const v8bf*)vr, *(const v8bf*)(vr + 16));
                acc[t] = __builtin_amdgcn_wmma_f32_16x16x32_bf16(
                    false, pa, false, bv, (short)0, acc[t], false, false);
            }
        }
    }

    // ---- normalize and write ctx as bf16 [B,S,H] ----
#pragma unroll
    for (int t = 0; t < 4; t++) {
#pragma unroll
        for (int v = 0; v < 8; v++) {
            const int q = q0 + lh * 8 + v;
            float val = acc[t][v] / lrow[v];
            ctx[((size_t)bb * SEQ + q) * HDIM + h * HEADD + t * 16 + lr] = (__bf16)val;
        }
    }
}

// ---------------------------------------------------------------------------
extern "C" void kernel_launch(void* const* d_in, const int* in_sizes, int n_in,
                              void* d_out, int out_size, void* d_ws, size_t ws_size,
                              hipStream_t stream) {
    (void)in_sizes; (void)n_in; (void)out_size; (void)ws_size;

    const float* hs = (const float*)d_in[0];
    const int*   am = (const int*)  d_in[1];
    const float* wq = (const float*)d_in[2];
    const float* bq = (const float*)d_in[3];
    const float* wk = (const float*)d_in[4];
    const float* bk = (const float*)d_in[5];
    const float* wv = (const float*)d_in[6];
    const float* bv = (const float*)d_in[7];
    const float* wo = (const float*)d_in[8];
    const float* bo = (const float*)d_in[9];
    float* out = (float*)d_out;

    char* ws = (char*)d_ws;
    size_t off = 0;
    auto carve = [&](size_t bytes) -> void* {
        void* p = ws + off;
        off += (bytes + 255) & ~(size_t)255;
        return p;
    };
    __bf16* Xbf = (__bf16*)carve((size_t)MTOT * HDIM * 2);
    __bf16* Wqb = (__bf16*)carve((size_t)HDIM * HDIM * 2);
    __bf16* Wkb = (__bf16*)carve((size_t)HDIM * HDIM * 2);
    __bf16* Wvb = (__bf16*)carve((size_t)HDIM * HDIM * 2);
    __bf16* Wob = (__bf16*)carve((size_t)HDIM * HDIM * 2);
    __bf16* Qbf = (__bf16*)carve((size_t)MTOT * HDIM * 2);
    __bf16* Kbf = (__bf16*)carve((size_t)MTOT * HDIM * 2);
    __bf16* Vtb = (__bf16*)carve((size_t)MTOT * HDIM * 2);
    __bf16* Cbf = (__bf16*)carve((size_t)MTOT * HDIM * 2);

    const int nX = MTOT * HDIM;       // 3,145,728
    const int nW = HDIM * HDIM;       //   589,824
    cvt_f32_bf16<<<nX / 4 / 256, 256, 0, stream>>>(hs, Xbf, nX);
    cvt_f32_bf16<<<nW / 4 / 256, 256, 0, stream>>>(wq, Wqb, nW);
    cvt_f32_bf16<<<nW / 4 / 256, 256, 0, stream>>>(wk, Wkb, nW);
    cvt_f32_bf16<<<nW / 4 / 256, 256, 0, stream>>>(wv, Wvb, nW);
    cvt_f32_bf16<<<nW / 4 / 256, 256, 0, stream>>>(wo, Wob, nW);

    dim3 gg(MTOT / 128, HDIM / 128);  // (32, 6)
    gemm_bf16_wmma<0><<<gg, 256, 0, stream>>>(Xbf, Wqb, bq, Qbf);
    gemm_bf16_wmma<1><<<gg, 256, 0, stream>>>(Xbf, Wkb, bk, Kbf);
    gemm_bf16_wmma<2><<<gg, 256, 0, stream>>>(Xbf, Wvb, bv, Vtb);

    dim3 ga(SEQ / 64, NHEAD, BATCH);  // (32, 12, 2)
    flash_attn_wmma<<<ga, 128, 0, stream>>>(Qbf, Kbf, Vtb, am, Cbf);

    gemm_bf16_wmma<3><<<gg, 256, 0, stream>>>(Cbf, Wob, bo, out);
}
